// DGCNNGraspQNet_61409442399006
// MI455X (gfx1250) — matrix operand order repaired and checked by reading, not compiled, not run on hardware
//
#include <hip/hip_runtime.h>
#include <cmath>

// ---------------------------------------------------------------------------
// Types for CDNA5 WMMA (wave32, 16x16x32 bf16 -> f32)
// ---------------------------------------------------------------------------
typedef __attribute__((ext_vector_type(16))) __bf16 v16bf;
typedef __attribute__((ext_vector_type(8)))  float  v8f;

struct Frag16 {
    union {
        v16bf    v;
        unsigned u[8];
    };
};

#define V8F_ZERO {0.f,0.f,0.f,0.f,0.f,0.f,0.f,0.f}

// bf16 <-> f32 helpers (round to nearest even)
__device__ __forceinline__ unsigned short f2bf(float x) {
    unsigned u = __float_as_uint(x);
    unsigned r = (u + 0x7FFFu + ((u >> 16) & 1u)) >> 16;
    return (unsigned short)r;
}
__device__ __forceinline__ float bf2f(unsigned short h) {
    return __uint_as_float(((unsigned)h) << 16);
}

// A fragment: 16x32 bf16, row-major source with leading dim lda (halfwords).
// ISA layout: lanes 0-15 rows M=0..15; VGPR g<4 -> K=2g, g>=4 -> K=16+2(g-4);
// lanes 16-31 shift K by +8.
__device__ __forceinline__ void load_a(Frag16& f, const unsigned short* A, int lda, int lane) {
    int row = lane & 15, half = lane >> 4;
    const unsigned short* base = A + row * lda;
#pragma unroll
    for (int g = 0; g < 8; ++g) {
        int k0 = ((g < 4) ? 0 : 16) + half * 8 + ((g & 3) << 1);
        f.u[g] = *reinterpret_cast<const unsigned*>(base + k0);
    }
}

// B fragment: 32x16 bf16. W stored [O][K] row-major (so B[k][n] = W[n][k] and the
// (k,k+1) pair is contiguous). lanes 0-15: N=0..15; K layout same as A (g<4 -> K=2g,
// g>=4 -> K=16+2(g-4); lanes 16-31 shift K by +8).
__device__ __forceinline__ void load_b(Frag16& f, const unsigned short* W, int ldw, int lane) {
    int col = lane & 15, khalf = lane >> 4;
    const unsigned short* base = W + col * ldw;
#pragma unroll
    for (int g = 0; g < 8; ++g)
        f.u[g] = *reinterpret_cast<const unsigned*>(base + ((g < 4) ? 0 : 16) + khalf * 8 + ((g & 3) << 1));
}

__device__ __forceinline__ v8f wmma_bf16(const Frag16& a, const Frag16& b, v8f c) {
    return __builtin_amdgcn_wmma_f32_16x16x32_bf16(
        /*neg_a=*/false, a.v, /*neg_b=*/false, b.v,
        /*c_mod=*/(short)0, c, /*reuse_a=*/false, /*reuse_b=*/false);
}

__device__ __forceinline__ void atomicMaxF(float* addr, float val) {
    unsigned* ua = (unsigned*)addr;
    unsigned old = *ua;
    while (true) {
        float f = __uint_as_float(old);
        if (f >= val) break;
        unsigned assumed = old;
        old = atomicCAS(ua, assumed, __float_as_uint(val));
        if (old == assumed) break;
    }
}

// ---------------------------------------------------------------------------
// Prep: f32 [O,K] -> bf16 [O,Kpad] (zero padded)
// ---------------------------------------------------------------------------
__global__ void convert_pad_kernel(const float* __restrict__ src, unsigned short* __restrict__ dst,
                                   int O, int K, int Kpad) {
    int i = blockIdx.x * 256 + threadIdx.x;
    if (i >= O * Kpad) return;
    int o = i / Kpad, k = i - o * Kpad;
    dst[i] = (k < K) ? f2bf(src[o * K + k]) : (unsigned short)0;
}

__global__ void init_pool_kernel(float* __restrict__ ofmax, float* __restrict__ ofsum, int n) {
    int i = blockIdx.x * 256 + threadIdx.x;
    if (i < n) { ofmax[i] = -3.0e38f; ofsum[i] = 0.f; }
}

// ---------------------------------------------------------------------------
// KNN: per-row top-20 of neg squared distance. One thread = one row.
// Block = 128 rows of one batch; column tiles of 128 staged through LDS.
// ---------------------------------------------------------------------------
__global__ __launch_bounds__(128) void knn_kernel(const unsigned short* __restrict__ X,
                                                  int D, int strideX, int N,
                                                  int* __restrict__ idxOut) {
    extern __shared__ float smf[];
    const int ldr = D + 1;                 // odd stride -> conflict-free per-lane reads
    float* rows = smf;                     // [128][ldr]
    float* cols = rows + 128 * ldr;        // [128][D]
    float* sqc  = cols + 128 * D;          // [128]

    int tilesPerB = N / 128;
    int b  = blockIdx.x / tilesPerB;
    int p0 = (blockIdx.x % tilesPerB) * 128;
    int t  = threadIdx.x;
    int n  = p0 + t;

    float sqr = 0.f;
    for (int d = 0; d < D; ++d) {
        float v = bf2f(X[(size_t)(b * N + n) * strideX + d]);
        rows[t * ldr + d] = v;
        sqr += v * v;
    }

    float vals[20];
    int   inds[20];
#pragma unroll
    for (int j = 0; j < 20; ++j) { vals[j] = -3.0e38f; inds[j] = 0; }

    for (int t0 = 0; t0 < N; t0 += 128) {
        __syncthreads();
        {
            int m = t0 + t;
            float s = 0.f;
            for (int d = 0; d < D; ++d) {
                float v = bf2f(X[(size_t)(b * N + m) * strideX + d]);
                cols[t * D + d] = v;
                s += v * v;
            }
            sqc[t] = s;
        }
        __syncthreads();
        for (int mm = 0; mm < 128; ++mm) {
            float dot = 0.f;
            const float* cr = cols + mm * D;
            const float* rr = rows + t * ldr;
            for (int d = 0; d < D; ++d) dot += rr[d] * cr[d];
            float negd = 2.f * dot - sqr - sqc[mm];
            if (negd > vals[19]) {
                int j = 19;
                while (j > 0 && vals[j - 1] < negd) {
                    vals[j] = vals[j - 1]; inds[j] = inds[j - 1]; --j;
                }
                vals[j] = negd; inds[j] = t0 + mm;
            }
        }
    }
#pragma unroll
    for (int j = 0; j < 20; ++j)
        idxOut[(size_t)(b * N + n) * 20 + j] = inds[j];
}

// ---------------------------------------------------------------------------
// EdgeConv: 8 points/block. E = [(x_nbr - x) || x], rows padded 20->32,
// WMMA GEMM against Wb[O][Kpad]. K-outer loop shares one B fragment across
// two independent accumulator chains (mt=0/1) so the XDL pipe can overlap.
// BN + lrelu + max-over-k folded into the C-fragment epilogue
// (per-lane max + shfl_xor(16) cross-half combine).
// ---------------------------------------------------------------------------
__global__ __launch_bounds__(256) void edgeconv_kernel(
    const unsigned short* __restrict__ X, int D, int strideX,
    const int* __restrict__ idx, const unsigned short* __restrict__ Wb, int Kpad,
    const float* __restrict__ gamma, const float* __restrict__ beta,
    unsigned short* __restrict__ Y, int strideY, int O, int N) {
    extern __shared__ unsigned short E[];          // [8][32][Kpad]
    __shared__ int sidx[8 * 20];

    int blocksPerB = N / 8;
    int b  = blockIdx.x / blocksPerB;
    int p0 = (blockIdx.x % blocksPerB) * 8;
    int twoD = 2 * D;

    for (int i = threadIdx.x; i < 160; i += 256)
        sidx[i] = idx[(size_t)(b * N + p0 + i / 20) * 20 + (i % 20)];
    __syncthreads();

    int total = 8 * 32 * Kpad;
    for (int e = threadIdx.x; e < total; e += 256) {
        int k  = e % Kpad;
        int rp = e / Kpad;
        int r  = rp & 31;
        int p  = rp >> 5;
        unsigned short val = 0;
        if (r < 20 && k < twoD) {
            int n  = p0 + p;
            float v;
            if (k < D) {
                float c = bf2f(X[(size_t)(b * N + n) * strideX + k]);
                int nb  = sidx[p * 20 + r];
                v = bf2f(X[(size_t)(b * N + nb) * strideX + k]) - c;
            } else {
                v = bf2f(X[(size_t)(b * N + n) * strideX + (k - D)]);
            }
            val = f2bf(v);
        }
        E[e] = val;
    }
    __syncthreads();

    int lane = threadIdx.x & 31, wave = threadIdx.x >> 5;
    int p = wave;                                  // 8 waves <-> 8 points
    float invs = rsqrtf(1.0f + 1e-5f);
    int nt_count = O >> 4;

    const unsigned short* A0 = E + (size_t)(p * 32) * Kpad;       // rows 0-15
    const unsigned short* A1 = E + (size_t)(p * 32 + 16) * Kpad;  // rows 16-31

    for (int nt = 0; nt < nt_count; ++nt) {
        v8f acc0 = V8F_ZERO;   // neighbor rows 0..15
        v8f acc1 = V8F_ZERO;   // neighbor rows 16..19 (+pad)
        for (int kc = 0; kc < Kpad; kc += 32) {
            Frag16 a0, a1, w;
            load_b(w,  Wb + (size_t)(nt * 16) * Kpad + kc, Kpad, lane);
            load_a(a0, A0 + kc, Kpad, lane);
            load_a(a1, A1 + kc, Kpad, lane);
            acc0 = wmma_bf16(a0, w, acc0);
            acc1 = wmma_bf16(a1, w, acc1);
        }
        int col  = nt * 16 + (lane & 15);
        float gv = gamma[col] * invs, bv = beta[col];
        float m  = -3.0e38f;
#pragma unroll
        for (int i = 0; i < 8; ++i) {               // rows 0..15 all valid
            float h = gv * acc0[i] + bv;
            h = (h >= 0.f) ? h : 0.2f * h;
            m = fmaxf(m, h);
        }
#pragma unroll
        for (int i = 0; i < 8; ++i) {               // rows 16..19 valid only
            int row = 16 + ((lane >> 4) << 3) + i;
            if (row < 20) {
                float h = gv * acc1[i] + bv;
                h = (h >= 0.f) ? h : 0.2f * h;
                m = fmaxf(m, h);
            }
        }
        m = fmaxf(m, __shfl_xor(m, 16, 32));
        if (lane < 16) {
            int n = p0 + p;
            Y[(size_t)(b * N + n) * strideY + col] = f2bf(m);
        }
    }
}

// ---------------------------------------------------------------------------
// Shared wave-level GEMM + bias + BN + lrelu, output to LDS (gripper layers)
// ---------------------------------------------------------------------------
__device__ __forceinline__ void wave_gemm_lds(
    const unsigned short* Ain, int lda, const unsigned short* Wb, int K,
    const float* bias, const float* gam, const float* bet,
    unsigned short* Out, int ldo, int Mtiles, int Ntiles, int lane, int wave) {
    float invs = rsqrtf(1.0f + 1e-5f);
    for (int tile = wave; tile < Mtiles * Ntiles; tile += 8) {
        int mt = tile / Ntiles, nt = tile - mt * Ntiles;
        v8f acc = V8F_ZERO;
        for (int kc = 0; kc < K; kc += 32) {
            Frag16 a, w;
            load_a(a, Ain + (size_t)(mt * 16) * lda + kc, lda, lane);
            load_b(w, Wb + (size_t)(nt * 16) * K + kc, K, lane);
            acc = wmma_bf16(a, w, acc);
        }
        int col  = nt * 16 + (lane & 15);
        float gv = gam[col] * invs, bv = bet[col], bi = bias[col];
#pragma unroll
        for (int i = 0; i < 8; ++i) {
            int row  = mt * 16 + ((lane >> 4) << 3) + i;
            float h  = (acc[i] + bi) * gv + bv;
            h = (h >= 0.f) ? h : 0.2f * h;
            Out[(size_t)row * ldo + col] = f2bf(h);
        }
    }
}

// ---------------------------------------------------------------------------
// Gripper encoder: MLP 3->64->128->128 per point, max over 256 points.
// ---------------------------------------------------------------------------
__global__ __launch_bounds__(256) void gripper_kernel(
    const float* __restrict__ grip,
    const float* __restrict__ gw1, const float* __restrict__ gb1,
    const float* __restrict__ gg1, const float* __restrict__ gbt1,
    const unsigned short* __restrict__ gw2b, const float* __restrict__ gb2,
    const float* __restrict__ gg2, const float* __restrict__ gbt2,
    const unsigned short* __restrict__ gw3b, const float* __restrict__ gb3,
    const float* __restrict__ gg3, const float* __restrict__ gbt3,
    float* __restrict__ gf) {
    extern __shared__ unsigned short sm[];
    unsigned short* bufA = sm;              // 256*128 halfwords
    unsigned short* bufB = sm + 256 * 128;  // 256*128 halfwords

    int b = blockIdx.x, t = threadIdx.x;
    float invs = rsqrtf(1.0f + 1e-5f);

    { // layer1: 3->64, plain VALU (K too small for WMMA to matter)
        const float* p = grip + (size_t)(b * 256 + t) * 3;
        float x0 = p[0], x1 = p[1], x2 = p[2];
        for (int o = 0; o < 64; ++o) {
            float s = gb1[o] + x0 * gw1[o * 3] + x1 * gw1[o * 3 + 1] + x2 * gw1[o * 3 + 2];
            s = gg1[o] * invs * s + gbt1[o];
            s = (s >= 0.f) ? s : 0.2f * s;
            bufA[t * 64 + o] = f2bf(s);
        }
    }
    __syncthreads();
    int lane = t & 31, wave = t >> 5;
    wave_gemm_lds(bufA, 64,  gw2b, 64,  gb2, gg2, gbt2, bufB, 128, 16, 8, lane, wave);
    __syncthreads();
    wave_gemm_lds(bufB, 128, gw3b, 128, gb3, gg3, gbt3, bufA, 128, 16, 8, lane, wave);
    __syncthreads();
    if (t < 128) {
        float m = -3.0e38f;
        for (int r = 0; r < 256; ++r) m = fmaxf(m, bf2f(bufA[r * 128 + t]));
        gf[b * 128 + t] = m;
    }
}

// ---------------------------------------------------------------------------
// x5: [1024,512] @ W5^T -> [1024,1024], BN+lrelu, fused max/sum pooling
// via atomics. Block tile 64x128, wave tile 32x32 (2x2 WMMA tiles).
// A tile staged global->LDS with GLOBAL_LOAD_ASYNC_TO_LDS_B128 (ASYNCcnt,
// CDNA5 ISA): 256 threads x 16B = full 64x32 bf16 tile,
// bypassing the VGPR round-trip.
// ---------------------------------------------------------------------------
__global__ __launch_bounds__(256) void x5_kernel(
    const unsigned short* __restrict__ xcat, const unsigned short* __restrict__ W5b,
    const float* __restrict__ g5, const float* __restrict__ b5,
    float* __restrict__ ofmax, float* __restrict__ ofsum) {
    __shared__ unsigned short As[64 * 32];

    int bi = blockIdx.x;
    int ct = bi & 7, rt = (bi >> 3) & 15, b = bi >> 7;
    int tid = threadIdx.x;
    int lane = tid & 31, wave = tid >> 5;
    int wr = wave >> 2, wc = wave & 3;

    // each thread owns one 16-byte chunk of the A tile per K step
    int arow = tid >> 2;                    // 0..63
    int akp  = (tid & 3) * 8;               // halfword offset within row chunk
    unsigned ldsAddr = (unsigned)(size_t)(&As[0]) + (unsigned)tid * 16u;
    const unsigned short* asrc_base =
        xcat + ((size_t)(b * 1024 + rt * 64 + arow)) * 512 + akp;

    v8f acc[2][2] = { {V8F_ZERO, V8F_ZERO}, {V8F_ZERO, V8F_ZERO} };

    for (int kc = 0; kc < 512; kc += 32) {
        __syncthreads();   // previous iteration's fragment reads are done
        {
            unsigned long long gaddr = (unsigned long long)(size_t)(asrc_base + kc);
            asm volatile("global_load_async_to_lds_b128 %0, %1, off"
                         :: "v"(ldsAddr), "v"(gaddr)
                         : "memory");
            asm volatile("s_wait_asynccnt 0x0" ::: "memory");
        }
        __syncthreads();   // whole tile visible to all waves
        Frag16 a0, a1, w0, w1;
        load_a(a0, As + (size_t)(wr * 32 + 0)  * 32, 32, lane);
        load_a(a1, As + (size_t)(wr * 32 + 16) * 32, 32, lane);
        load_b(w0, W5b + (size_t)(ct * 128 + wc * 32 + 0)  * 512 + kc, 512, lane);
        load_b(w1, W5b + (size_t)(ct * 128 + wc * 32 + 16) * 512 + kc, 512, lane);
        acc[0][0] = wmma_bf16(a0, w0, acc[0][0]);
        acc[0][1] = wmma_bf16(a0, w1, acc[0][1]);
        acc[1][0] = wmma_bf16(a1, w0, acc[1][0]);
        acc[1][1] = wmma_bf16(a1, w1, acc[1][1]);
    }

    float invs = rsqrtf(1.0f + 1e-5f);
#pragma unroll
    for (int nt = 0; nt < 2; ++nt) {
        int col  = ct * 128 + wc * 32 + nt * 16 + (lane & 15);
        float gv = g5[col] * invs, bv = b5[col];
        float m = -3.0e38f, s = 0.f;
#pragma unroll
        for (int mt = 0; mt < 2; ++mt) {
#pragma unroll
            for (int i = 0; i < 8; ++i) {
                float h = gv * acc[mt][nt][i] + bv;
                h = (h >= 0.f) ? h : 0.2f * h;
                m = fmaxf(m, h);
                s += h;
            }
        }
        m = fmaxf(m, __shfl_xor(m, 16, 32));
        s = s + __shfl_xor(s, 16, 32);
        if (lane < 16) {
            atomicMaxF(&ofmax[b * 1024 + col], m);
            atomicAdd(&ofsum[b * 1024 + col], s);
        }
    }
}

// ---------------------------------------------------------------------------
// Fusion + score head: [2176]->512->512->256->1 + sigmoid. Tiny: VALU.
// ---------------------------------------------------------------------------
__global__ __launch_bounds__(256) void head_kernel(
    const float* __restrict__ gf, const float* __restrict__ ofmax, const float* __restrict__ ofsum,
    const float* __restrict__ fw,  const float* __restrict__ fb,
    const float* __restrict__ fg,  const float* __restrict__ fbt,
    const float* __restrict__ sw1, const float* __restrict__ sb1,
    const float* __restrict__ sg1, const float* __restrict__ sbt1,
    const float* __restrict__ sw2, const float* __restrict__ sb2,
    const float* __restrict__ sg2, const float* __restrict__ sbt2,
    const float* __restrict__ sw3, const float* __restrict__ sb3,
    float* __restrict__ out) {
    __shared__ float z[2176];
    __shared__ float h1[512];
    __shared__ float h2[512];
    __shared__ float h3[256];
    int b = blockIdx.x, t = threadIdx.x;
    float invs = rsqrtf(1.0f + 1e-5f);

    for (int i = t; i < 128;  i += 256) z[i]        = gf[b * 128 + i];
    for (int i = t; i < 1024; i += 256) z[128 + i]  = ofmax[b * 1024 + i];
    for (int i = t; i < 1024; i += 256) z[1152 + i] = ofsum[b * 1024 + i] * (1.0f / 1024.0f);
    __syncthreads();

    for (int o = t; o < 512; o += 256) {
        float s = fb[o];
        const float* w = fw + (size_t)o * 2176;
        for (int i = 0; i < 2176; ++i) s += z[i] * w[i];
        s = fg[o] * invs * s + fbt[o];
        h1[o] = (s >= 0.f) ? s : 0.2f * s;
    }
    __syncthreads();
    for (int o = t; o < 512; o += 256) {
        float s = sb1[o];
        const float* w = sw1 + (size_t)o * 512;
        for (int i = 0; i < 512; ++i) s += h1[i] * w[i];
        s = sg1[o] * invs * s + sbt1[o];
        h2[o] = (s >= 0.f) ? s : 0.2f * s;
    }
    __syncthreads();
    for (int o = t; o < 256; o += 256) {
        float s = sb2[o];
        const float* w = sw2 + (size_t)o * 512;
        for (int i = 0; i < 512; ++i) s += h2[i] * w[i];
        s = sg2[o] * invs * s + sbt2[o];
        h3[o] = (s >= 0.f) ? s : 0.2f * s;
    }
    __syncthreads();
    if (t == 0) {
        float s = sb3[0];
        for (int i = 0; i < 256; ++i) s += h3[i] * sw3[i];
        out[b] = 1.0f / (1.0f + expf(-s));
    }
}

// ---------------------------------------------------------------------------
// Host launcher
// ---------------------------------------------------------------------------
extern "C" void kernel_launch(void* const* d_in, const int* in_sizes, int n_in,
                              void* d_out, int out_size, void* d_ws, size_t ws_size,
                              hipStream_t stream) {
    (void)in_sizes; (void)n_in; (void)out_size; (void)ws_size;
    const int B = 16, N = 1024;

    const float* obj  = (const float*)d_in[0];
    const float* grip = (const float*)d_in[1];
    const float* W1 = (const float*)d_in[2],  *g1 = (const float*)d_in[3],  *b1 = (const float*)d_in[4];
    const float* W2 = (const float*)d_in[5],  *g2 = (const float*)d_in[6],  *b2 = (const float*)d_in[7];
    const float* W3 = (const float*)d_in[8],  *g3 = (const float*)d_in[9],  *b3 = (const float*)d_in[10];
    const float* W4 = (const float*)d_in[11], *g4 = (const float*)d_in[12], *b4 = (const float*)d_in[13];
    const float* W5 = (const float*)d_in[14], *g5 = (const float*)d_in[15], *b5 = (const float*)d_in[16];
    const float* gw1 = (const float*)d_in[17], *gb1 = (const float*)d_in[18];
    const float* gg1 = (const float*)d_in[19], *gbt1 = (const float*)d_in[20];
    const float* gw2 = (const float*)d_in[21], *gb2 = (const float*)d_in[22];
    const float* gg2 = (const float*)d_in[23], *gbt2 = (const float*)d_in[24];
    const float* gw3 = (const float*)d_in[25], *gb3 = (const float*)d_in[26];
    const float* gg3 = (const float*)d_in[27], *gbt3 = (const float*)d_in[28];
    const float* fw  = (const float*)d_in[29], *fb  = (const float*)d_in[30];
    const float* fg  = (const float*)d_in[31], *fbt = (const float*)d_in[32];
    const float* sw1 = (const float*)d_in[33], *sb1 = (const float*)d_in[34];
    const float* sg1 = (const float*)d_in[35], *sbt1 = (const float*)d_in[36];
    const float* sw2 = (const float*)d_in[37], *sb2 = (const float*)d_in[38];
    const float* sg2 = (const float*)d_in[39], *sbt2 = (const float*)d_in[40];
    const float* sw3 = (const float*)d_in[41], *sb3 = (const float*)d_in[42];
    float* out = (float*)d_out;

    char* ws = (char*)d_ws;
    size_t off = 0;
    auto wsoff = [&](size_t bytes) -> char* {
        char* p = ws + off;
        off += (bytes + 255) & ~(size_t)255;
        return p;
    };
    unsigned short* objb  = (unsigned short*)wsoff((size_t)B * N * 4 * 2);
    unsigned short* W1b   = (unsigned short*)wsoff((size_t)64 * 32 * 2);
    unsigned short* W2b   = (unsigned short*)wsoff((size_t)64 * 128 * 2);
    unsigned short* W3b   = (unsigned short*)wsoff((size_t)128 * 128 * 2);
    unsigned short* W4b   = (unsigned short*)wsoff((size_t)256 * 256 * 2);
    unsigned short* W5b   = (unsigned short*)wsoff((size_t)1024 * 512 * 2);
    unsigned short* gw2b  = (unsigned short*)wsoff((size_t)128 * 64 * 2);
    unsigned short* gw3b  = (unsigned short*)wsoff((size_t)128 * 128 * 2);
    int*            idxb  = (int*)wsoff((size_t)B * N * 20 * 4);
    unsigned short* xcat  = (unsigned short*)wsoff((size_t)B * N * 512 * 2);
    float*          gfbuf = (float*)wsoff((size_t)B * 128 * 4);
    float*          ofmax = (float*)wsoff((size_t)B * 1024 * 4);
    float*          ofsum = (float*)wsoff((size_t)B * 1024 * 4);

    auto cvt = [&](const float* s, unsigned short* d, int O, int K, int Kp) {
        int tot = O * Kp;
        convert_pad_kernel<<<(tot + 255) / 256, 256, 0, stream>>>(s, d, O, K, Kp);
    };
    cvt(W1, W1b, 64, 6, 32);
    cvt(W2, W2b, 64, 128, 128);
    cvt(W3, W3b, 128, 128, 128);
    cvt(W4, W4b, 256, 256, 256);
    cvt(W5, W5b, 1024, 512, 512);
    cvt(gw2, gw2b, 128, 64, 64);
    cvt(gw3, gw3b, 128, 128, 128);
    cvt(obj, objb, B * N, 3, 4);

    init_pool_kernel<<<(B * 1024 + 255) / 256, 256, 0, stream>>>(ofmax, ofsum, B * 1024);

    gripper_kernel<<<B, 256, 2 * 256 * 128 * 2, stream>>>(
        grip, gw1, gb1, gg1, gbt1, gw2b, gb2, gg2, gbt2, gw3b, gb3, gg3, gbt3, gfbuf);

    auto knn_sm = [&](int D) { return (size_t)(128 * (D + 1) + 128 * D + 128) * sizeof(float); };
    auto ec_sm  = [&](int Kp) { return (size_t)8 * 32 * Kp * 2; };
    int knnBlocks = B * (N / 128);
    int ecBlocks  = B * (N / 8);

    // stage 1: knn on coords, conv 6->64 -> xcat[:,0:64]
    knn_kernel<<<knnBlocks, 128, knn_sm(3), stream>>>(objb, 3, 4, N, idxb);
    edgeconv_kernel<<<ecBlocks, 256, ec_sm(32), stream>>>(
        objb, 3, 4, idxb, W1b, 32, g1, b1, xcat + 0, 512, 64, N);

    // stage 2: knn on x1 (64), conv 128->64 -> xcat[:,64:128]
    knn_kernel<<<knnBlocks, 128, knn_sm(64), stream>>>(xcat + 0, 64, 512, N, idxb);
    edgeconv_kernel<<<ecBlocks, 256, ec_sm(128), stream>>>(
        xcat + 0, 64, 512, idxb, W2b, 128, g2, b2, xcat + 64, 512, 64, N);

    // stage 3: knn on x2 (64), conv 128->128 -> xcat[:,128:256]
    knn_kernel<<<knnBlocks, 128, knn_sm(64), stream>>>(xcat + 64, 64, 512, N, idxb);
    edgeconv_kernel<<<ecBlocks, 256, ec_sm(128), stream>>>(
        xcat + 64, 64, 512, idxb, W3b, 128, g3, b3, xcat + 128, 512, 128, N);

    // stage 4: knn on x3 (128), conv 256->256 -> xcat[:,256:512]
    knn_kernel<<<knnBlocks, 128, knn_sm(128), stream>>>(xcat + 128, 128, 512, N, idxb);
    edgeconv_kernel<<<ecBlocks, 256, ec_sm(256), stream>>>(
        xcat + 128, 128, 512, idxb, W4b, 256, g4, b4, xcat + 256, 512, 256, N);

    // x5 GEMM + fused global max/mean pooling
    x5_kernel<<<B * 16 * 8, 256, 0, stream>>>(xcat, W5b, g5, b5, ofmax, ofsum);

    // fusion + score head
    head_kernel<<<B, 256, 0, stream>>>(gfbuf, ofmax, ofsum,
                                       fw, fb, fg, fbt,
                                       sw1, sb1, sg1, sbt1,
                                       sw2, sb2, sg2, sbt2,
                                       sw3, sb3, out);
}
